// GraphLinkPredictor_86208583565975
// MI455X (gfx1250) — matrix operand order repaired and checked by reading, not compiled, not run on hardware
//
#include <hip/hip_runtime.h>
#include <hip/hip_bf16.h>

// ---------------------------------------------------------------------------
// GraphLinkPredictor for MI455X (gfx1250): bf16 WMMA GEMMs + atomic GAT
// softmax + algebraic refactor of the link predictor (concat-GEMM split into
// two per-node GEMMs, pair stage becomes bandwidth-bound gather+dot).
// ---------------------------------------------------------------------------

typedef unsigned short u16;
typedef __attribute__((ext_vector_type(16))) __bf16 v16bf;
typedef __attribute__((ext_vector_type(8)))  __bf16 v8bf;
typedef __attribute__((ext_vector_type(8)))  float  v8f;

#define CDIV(a, b) (((a) + (b) - 1) / (b))

__device__ __forceinline__ u16 f32_to_bf16(float f) {
    unsigned int u = __float_as_uint(f);
    u += 0x7FFFu + ((u >> 16) & 1u);   // round-to-nearest-even
    return (u16)(u >> 16);
}

__device__ __forceinline__ void atomicMaxFloat(float* addr, float val) {
    if (val >= 0.0f) atomicMax((int*)addr, __float_as_int(val));
    else             atomicMin((unsigned int*)addr, __float_as_uint(val));
}

// ---------------------------------------------------------------------------
// Elementwise helpers
// ---------------------------------------------------------------------------
__global__ __launch_bounds__(256) void k_fill_f32(float* p, float v, int n) {
    int i = blockIdx.x * 256 + threadIdx.x;
    if (i < n) p[i] = v;
}

__global__ __launch_bounds__(256) void k_cvt_f32_bf16(const float* __restrict__ s,
                                                      u16* __restrict__ d, int n) {
    int i = blockIdx.x * 256 + threadIdx.x;
    if (i < n) d[i] = f32_to_bf16(s[i]);
}

// ---------------------------------------------------------------------------
// bf16 WMMA GEMM: C[M,N] = A[M,K] (bf16 rm) * B[K,N] (bf16 rm), f32 out.
// 256 threads = 8 waves, 64x64 tile, K-step 32, LDS staged (B transposed).
// Requires: K % 32 == 0, N % 64 == 0. M guarded.
// ---------------------------------------------------------------------------
__global__ __launch_bounds__(256) void k_gemm_bf16_wmma(
    const u16* __restrict__ A, const u16* __restrict__ B, float* __restrict__ C,
    int M, int N, int K) {
    constexpr int TM = 64, TN = 64, TK = 32;
    __shared__ __attribute__((aligned(16))) u16 lA[TM * TK];   // [row][k]
    __shared__ __attribute__((aligned(16))) u16 lBt[TN * TK];  // [col][k]

    const int bm = blockIdx.y * TM;
    const int bn = blockIdx.x * TN;
    const int tid = threadIdx.x;
    const int wave = tid >> 5;
    const int lane = tid & 31;
    const int half = lane >> 4;      // lane-half selects K sub-range
    const int l15  = lane & 15;
    const int wm = (wave >> 1) * 16; // wave row base inside tile (0..48)
    const int wn = (wave & 1) * 32;  // wave col base inside tile (0 or 32)

    v8f acc0 = {0.f, 0.f, 0.f, 0.f, 0.f, 0.f, 0.f, 0.f};
    v8f acc1 = {0.f, 0.f, 0.f, 0.f, 0.f, 0.f, 0.f, 0.f};

    for (int k0 = 0; k0 < K; k0 += TK) {
        // ---- stage A tile: 64x32, 8 contiguous bf16 per thread (b128) ----
        {
            int row = tid >> 2;          // 0..63
            int ko  = (tid & 3) * 8;     // 0,8,16,24
            int gr  = bm + row;
            uint4 t = {0u, 0u, 0u, 0u};
            if (gr < M) t = *(const uint4*)(A + (size_t)gr * K + k0 + ko);
            *(uint4*)&lA[row * TK + ko] = t;
        }
        // ---- stage B tile transposed: global [32k x 64n] -> lBt[n][k] ----
        {
            int k  = tid >> 3;           // 0..31
            int n0 = (tid & 7) * 8;      // 0..56
            union { uint4 v; u16 u[8]; } t;
            t.v = *(const uint4*)(B + (size_t)(k0 + k) * N + bn + n0);
#pragma unroll
            for (int i = 0; i < 8; ++i) lBt[(n0 + i) * TK + k] = t.u[i];
        }
        __syncthreads();

        // ---- build fragments (contiguous k order: k = 16*half + element, same for A and B) ----
        const int kbA = half * 16;       // lanes 0-15: K0-15, lanes 16-31: K16-31
        const u16* ap = &lA[(wm + l15) * TK + kbA];
        v8bf alo = *(const v8bf*)ap;
        v8bf ahi = *(const v8bf*)(ap + 8);
        v16bf af;
#pragma unroll
        for (int i = 0; i < 8; ++i) { af[i] = alo[i]; af[i + 8] = ahi[i]; }

        const int kbB = half * 16;       // lanes 0-15: K0-15, lanes 16-31: K16-31
        const u16* bp0 = &lBt[(wn + l15) * TK + kbB];
        const u16* bp1 = &lBt[(wn + 16 + l15) * TK + kbB];
        v8bf b0lo = *(const v8bf*)bp0, b0hi = *(const v8bf*)(bp0 + 8);
        v8bf b1lo = *(const v8bf*)bp1, b1hi = *(const v8bf*)(bp1 + 8);
        v16bf bf0, bf1v;
#pragma unroll
        for (int i = 0; i < 8; ++i) {
            bf0[i] = b0lo[i]; bf0[i + 8] = b0hi[i];
            bf1v[i] = b1lo[i]; bf1v[i + 8] = b1hi[i];
        }

        acc0 = __builtin_amdgcn_wmma_f32_16x16x32_bf16(false, af, false, bf0,
                                                       (short)0, acc0, false, false);
        acc1 = __builtin_amdgcn_wmma_f32_16x16x32_bf16(false, af, false, bf1v,
                                                       (short)0, acc1, false, false);
        __syncthreads();
    }

    // ---- store C: lane holds rows half*8+v, col l15 (ISA C layout) ----
    const int gcol = bn + wn + l15;
#pragma unroll
    for (int v = 0; v < 8; ++v) {
        int gr = bm + wm + half * 8 + v;
        if (gr < M) {
            C[(size_t)gr * N + gcol]      = acc0[v];
            C[(size_t)gr * N + gcol + 16] = acc1[v];
        }
    }
}

// ---------------------------------------------------------------------------
// GAT pieces
// ---------------------------------------------------------------------------
// es[n,h] = sum_c h[n,h,c]*a_s[h,c]; ed likewise
__global__ __launch_bounds__(256) void k_gat_coef(
    const float* __restrict__ h, const float* __restrict__ a_s,
    const float* __restrict__ a_d, float* __restrict__ es, float* __restrict__ ed,
    int n, int heads, int ch) {
    int i = blockIdx.x * 256 + threadIdx.x;
    if (i >= n * heads) return;
    int node = i / heads, hd = i % heads;
    const float* hp = h + (size_t)node * heads * ch + (size_t)hd * ch;
    const float* as = a_s + hd * ch;
    const float* ad = a_d + hd * ch;
    float ss = 0.f, sd = 0.f;
    for (int c = 0; c < ch; ++c) { ss += hp[c] * as[c]; sd += hp[c] * ad[c]; }
    es[i] = ss;
    ed[i] = sd;
}

// pass 1: el = leaky_relu(es[src]+ed[dst]); atomicMax per (dst,head)
__global__ __launch_bounds__(256) void k_edge_logit_max(
    const int* __restrict__ ei, int E, int nNodes, int heads,
    const float* __restrict__ es, const float* __restrict__ ed,
    float* __restrict__ el, float* __restrict__ m) {
    int idx = blockIdx.x * 256 + threadIdx.x;
    int total = (E + nNodes) * heads;
    if (idx >= total) return;
    int e = idx / heads, hd = idx % heads;
    int s, d;
    if (e < E) { s = ei[e]; d = ei[E + e]; } else { s = d = e - E; }  // self loops
    float v = es[s * heads + hd] + ed[d * heads + hd];
    v = v > 0.f ? v : 0.2f * v;
    el[idx] = v;
    atomicMaxFloat(&m[d * heads + hd], v);
}

// pass 2: ex = exp(el - m[dst]) (in place); atomicAdd into s[dst,head]
__global__ __launch_bounds__(256) void k_edge_exp_sum(
    const int* __restrict__ ei, int E, int nNodes, int heads,
    const float* __restrict__ m, float* __restrict__ el, float* __restrict__ ssum) {
    int idx = blockIdx.x * 256 + threadIdx.x;
    int total = (E + nNodes) * heads;
    if (idx >= total) return;
    int e = idx / heads, hd = idx % heads;
    int d = (e < E) ? ei[E + e] : (e - E);
    float ex = __expf(el[idx] - m[d * heads + hd]);
    el[idx] = ex;
    atomicAdd(&ssum[d * heads + hd], ex);
}

// pass 3: out[dst] += (ex/s[dst]) * h[src], 4 channels per thread
__global__ __launch_bounds__(256) void k_edge_aggregate(
    const int* __restrict__ ei, int E, int nNodes, int heads, int ch,
    const float* __restrict__ ex, const float* __restrict__ ssum,
    const float* __restrict__ h, float* __restrict__ out) {
    int qc = heads * (ch >> 2);                  // quads per edge
    long long idx = (long long)blockIdx.x * 256 + threadIdx.x;
    long long total = (long long)(E + nNodes) * qc;
    if (idx >= total) return;
    int e = (int)(idx / qc);
    int q = (int)(idx % qc);
    int hd = q / (ch >> 2);
    int c4 = (q % (ch >> 2)) * 4;
    int s, d;
    if (e < E) { s = ei[e]; d = ei[E + e]; } else { s = d = e - E; }
    float w = ex[e * heads + hd] / (ssum[d * heads + hd] + 1e-16f);
    const float4 hv = *(const float4*)(h + (size_t)s * heads * ch + (size_t)hd * ch + c4);
    float* op = out + (size_t)d * heads * ch + (size_t)hd * ch + c4;
    atomicAdd(op + 0, w * hv.x);
    atomicAdd(op + 1, w * hv.y);
    atomicAdd(op + 2, w * hv.z);
    atomicAdd(op + 3, w * hv.w);
}

// ---------------------------------------------------------------------------
// y = act(LN(x + bias; g, be)) -> bf16.  One block (256 thr) per row.
// act: 0=none, 1=relu, 2=elu
// ---------------------------------------------------------------------------
__global__ __launch_bounds__(256) void k_bias_ln_act_bf16(
    const float* __restrict__ in, const float* __restrict__ bias,
    const float* __restrict__ g, const float* __restrict__ be,
    u16* __restrict__ outb, int D, int act) {
    __shared__ float red[256];
    int row = blockIdx.x;
    const float* x = in + (size_t)row * D;
    float s = 0.f;
    for (int i = threadIdx.x; i < D; i += 256) s += x[i] + bias[i];
    red[threadIdx.x] = s;
    __syncthreads();
    for (int o = 128; o > 0; o >>= 1) {
        if (threadIdx.x < o) red[threadIdx.x] += red[threadIdx.x + o];
        __syncthreads();
    }
    float mean = red[0] / (float)D;
    __syncthreads();
    float v = 0.f;
    for (int i = threadIdx.x; i < D; i += 256) {
        float dlt = x[i] + bias[i] - mean;
        v += dlt * dlt;
    }
    red[threadIdx.x] = v;
    __syncthreads();
    for (int o = 128; o > 0; o >>= 1) {
        if (threadIdx.x < o) red[threadIdx.x] += red[threadIdx.x + o];
        __syncthreads();
    }
    float inv = rsqrtf(red[0] / (float)D + 1e-5f);
    for (int i = threadIdx.x; i < D; i += 256) {
        float y = (x[i] + bias[i] - mean) * inv * g[i] + be[i];
        if (act == 1)      y = fmaxf(y, 0.f);
        else if (act == 2) y = y > 0.f ? y : (__expf(y) - 1.f);
        outb[(size_t)row * D + i] = f32_to_bf16(y);
    }
}

// fb[:,0:768] = bf16(proj + bp); fb[:,768:1536] = bf16(bert)
__global__ __launch_bounds__(256) void k_concat_proj_bert(
    const float* __restrict__ proj, const float* __restrict__ bp,
    const float* __restrict__ bert, u16* __restrict__ fb, int n) {
    long long idx = (long long)blockIdx.x * 256 + threadIdx.x;
    if (idx >= (long long)n * 1536) return;
    int row = (int)(idx / 1536);
    int c = (int)(idx % 1536);
    float v = (c < 768) ? (proj[(size_t)row * 768 + c] + bp[c])
                        : bert[(size_t)row * 768 + (c - 768)];
    fb[idx] = f32_to_bf16(v);
}

// out[p] = relu(A[a]+B[b]+bl1) . Wl2 + bl2   -- one wave per pair
__global__ __launch_bounds__(256) void k_pair_predict(
    const int* __restrict__ pe, int P, const float* __restrict__ Abuf,
    const float* __restrict__ Bbuf, const float* __restrict__ bl1,
    const float* __restrict__ wl2, const float* __restrict__ bl2,
    float* __restrict__ out) {
    int wid = (blockIdx.x * 256 + threadIdx.x) >> 5;
    int lane = threadIdx.x & 31;
    if (wid >= P) return;
    int a = pe[wid], b = pe[P + wid];
    const float* ar = Abuf + (size_t)a * 768;
    const float* br = Bbuf + (size_t)b * 768;
    float acc = 0.f;
    for (int i = lane; i < 768; i += 32) {
        float v = ar[i] + br[i] + bl1[i];
        v = fmaxf(v, 0.f);
        acc += v * wl2[i];
    }
#pragma unroll
    for (int off = 16; off > 0; off >>= 1) acc += __shfl_down(acc, off, 32);
    if (lane == 0) out[wid] = acc + bl2[0];
}

// ---------------------------------------------------------------------------
// Orchestration
// ---------------------------------------------------------------------------
extern "C" void kernel_launch(void* const* d_in, const int* in_sizes, int n_in,
                              void* d_out, int out_size, void* d_ws, size_t ws_size,
                              hipStream_t stream) {
    (void)in_sizes; (void)n_in; (void)out_size; (void)ws_size;
    constexpr int N = 10000, E = 160000, P = 100000;
    constexpr int IN = 256, HID = 128, NH = 4, OUT = 256, H2 = 2, BH = 768, FH = 512;
    const int EP = E + N;

    const float* x    = (const float*)d_in[0];
    const int*   ei   = (const int*)d_in[1];
    const int*   pe   = (const int*)d_in[2];
    const float* bert = (const float*)d_in[3];
    const float* W1   = (const float*)d_in[4];
    const float* a1s  = (const float*)d_in[5];
    const float* a1d  = (const float*)d_in[6];
    const float* b1   = (const float*)d_in[7];
    const float* g1   = (const float*)d_in[8];
    const float* be1  = (const float*)d_in[9];
    const float* W2   = (const float*)d_in[10];
    const float* a2s  = (const float*)d_in[11];
    const float* a2d  = (const float*)d_in[12];
    const float* b2   = (const float*)d_in[13];
    const float* g2   = (const float*)d_in[14];
    const float* be2  = (const float*)d_in[15];
    const float* Wp   = (const float*)d_in[16];
    const float* bp   = (const float*)d_in[17];
    const float* Wf1  = (const float*)d_in[18];
    const float* bf1  = (const float*)d_in[19];
    const float* gf1  = (const float*)d_in[20];
    const float* bef1 = (const float*)d_in[21];
    const float* Wf2  = (const float*)d_in[22];
    const float* bf2  = (const float*)d_in[23];
    const float* gf2  = (const float*)d_in[24];
    const float* bef2 = (const float*)d_in[25];
    const float* Wl1  = (const float*)d_in[26];
    const float* bl1  = (const float*)d_in[27];
    const float* Wl2  = (const float*)d_in[28];
    const float* bl2  = (const float*)d_in[29];
    float* out = (float*)d_out;

    // ---- workspace: 4 ping-pong 32MB slots + small bump region ----
    char* ws = (char*)d_ws;
    const size_t SLOT = 32ull << 20;
    char* s0 = ws;
    char* s1 = ws + SLOT;
    char* s2 = ws + 2 * SLOT;
    char* s3 = ws + 3 * SLOT;
    char* sm = ws + 4 * SLOT;
    auto bump = [&](size_t bytes) {
        char* p = sm;
        sm += (bytes + 255) & ~(size_t)255;
        return p;
    };
    u16* W1b  = (u16*)bump((size_t)IN * NH * HID * 2);
    u16* W2b  = (u16*)bump((size_t)NH * HID * H2 * OUT * 2);
    u16* Wpb  = (u16*)bump((size_t)H2 * OUT * BH * 2);
    u16* Wf1b = (u16*)bump((size_t)2 * BH * FH * 2);
    u16* Wf2b = (u16*)bump((size_t)FH * BH * 2);
    u16* Wl1b = (u16*)bump((size_t)2 * BH * BH * 2);
    float* es1 = (float*)bump((size_t)N * NH * 4);
    float* ed1 = (float*)bump((size_t)N * NH * 4);
    float* m1  = (float*)bump((size_t)N * NH * 4);
    float* ss1 = (float*)bump((size_t)N * NH * 4);
    float* el1 = (float*)bump((size_t)EP * NH * 4);
    float* es2 = (float*)bump((size_t)N * H2 * 4);
    float* ed2 = (float*)bump((size_t)N * H2 * 4);
    float* m2  = (float*)bump((size_t)N * H2 * 4);
    float* ss2 = (float*)bump((size_t)N * H2 * 4);
    float* el2 = (float*)bump((size_t)EP * H2 * 4);

    // slot lifetimes (verified non-overlapping):
    u16*   xb    = (u16*)s0;    // s0: xb -> agg1 -> z1 -> Abuf
    float* agg1  = (float*)s0;
    float* z1    = (float*)s0;
    float* Abuf  = (float*)s0;
    float* h1    = (float*)s1;  // s1: h1 -> x2b -> fb -> z2 -> Bbuf
    u16*   x2b   = (u16*)s1;
    u16*   fb    = (u16*)s1;
    float* z2    = (float*)s1;
    float* Bbuf  = (float*)s1;
    float* h2    = (float*)s2;  // s2: h2 -> gnnb -> z1b -> neb
    u16*   gnnb  = (u16*)s2;
    u16*   z1b   = (u16*)s2;
    u16*   neb   = (u16*)s2;
    float* agg2  = (float*)s3;  // s3: agg2 -> projf
    float* projf = (float*)s3;

    auto gemm = [&](const u16* A, const u16* B, float* C, int M, int Nc, int K) {
        dim3 grid(Nc / 64, CDIV(M, 64));
        k_gemm_bf16_wmma<<<grid, 256, 0, stream>>>(A, B, C, M, Nc, K);
    };
    auto cvt = [&](const float* s, u16* d, size_t n) {
        k_cvt_f32_bf16<<<CDIV((int)n, 256), 256, 0, stream>>>(s, d, (int)n);
    };
    auto fill = [&](float* p, float v, int n) {
        k_fill_f32<<<CDIV(n, 256), 256, 0, stream>>>(p, v, n);
    };

    // weight conversions (f32 -> bf16)
    cvt(W1,  W1b,  (size_t)IN * NH * HID);
    cvt(W2,  W2b,  (size_t)NH * HID * H2 * OUT);
    cvt(Wp,  Wpb,  (size_t)H2 * OUT * BH);
    cvt(Wf1, Wf1b, (size_t)2 * BH * FH);
    cvt(Wf2, Wf2b, (size_t)FH * BH);
    cvt(Wl1, Wl1b, (size_t)2 * BH * BH);
    cvt(x, xb, (size_t)N * IN);

    // ---------------- GAT1: 256 -> 4x128 ----------------
    gemm(xb, W1b, h1, N, NH * HID, IN);                                     // h1 = x@W1
    k_gat_coef<<<CDIV(N * NH, 256), 256, 0, stream>>>(h1, a1s, a1d, es1, ed1, N, NH, HID);
    fill(m1, -1e30f, N * NH);
    fill(ss1, 0.f, N * NH);
    k_edge_logit_max<<<CDIV(EP * NH, 256), 256, 0, stream>>>(ei, E, N, NH, es1, ed1, el1, m1);
    k_edge_exp_sum<<<CDIV(EP * NH, 256), 256, 0, stream>>>(ei, E, N, NH, m1, el1, ss1);
    fill(agg1, 0.f, N * NH * HID);
    {
        long long tot = (long long)EP * NH * (HID / 4);
        k_edge_aggregate<<<(int)CDIV(tot, 256), 256, 0, stream>>>(ei, E, N, NH, HID,
                                                                  el1, ss1, h1, agg1);
    }
    k_bias_ln_act_bf16<<<N, 256, 0, stream>>>(agg1, b1, g1, be1, x2b, NH * HID, /*elu*/2);

    // ---------------- GAT2: 512 -> 2x256 ----------------
    gemm(x2b, W2b, h2, N, H2 * OUT, NH * HID);
    k_gat_coef<<<CDIV(N * H2, 256), 256, 0, stream>>>(h2, a2s, a2d, es2, ed2, N, H2, OUT);
    fill(m2, -1e30f, N * H2);
    fill(ss2, 0.f, N * H2);
    k_edge_logit_max<<<CDIV(EP * H2, 256), 256, 0, stream>>>(ei, E, N, H2, es2, ed2, el2, m2);
    k_edge_exp_sum<<<CDIV(EP * H2, 256), 256, 0, stream>>>(ei, E, N, H2, m2, el2, ss2);
    fill(agg2, 0.f, N * H2 * OUT);
    {
        long long tot = (long long)EP * H2 * (OUT / 4);
        k_edge_aggregate<<<(int)CDIV(tot, 256), 256, 0, stream>>>(ei, E, N, H2, OUT,
                                                                  el2, ss2, h2, agg2);
    }
    k_bias_ln_act_bf16<<<N, 256, 0, stream>>>(agg2, b2, g2, be2, gnnb, H2 * OUT, /*elu*/2);

    // ---------------- fusion ----------------
    gemm(gnnb, Wpb, projf, N, BH, H2 * OUT);                                // proj (bias in concat)
    k_concat_proj_bert<<<(int)CDIV((long long)N * 1536, 256), 256, 0, stream>>>(
        projf, bp, bert, fb, N);
    gemm(fb, Wf1b, z1, N, FH, 2 * BH);                                      // f@Wf1
    k_bias_ln_act_bf16<<<N, 256, 0, stream>>>(z1, bf1, gf1, bef1, z1b, FH, /*relu*/1);
    gemm(z1b, Wf2b, z2, N, BH, FH);                                         // @Wf2
    k_bias_ln_act_bf16<<<N, 256, 0, stream>>>(z2, bf2, gf2, bef2, neb, BH, /*none*/0);

    // ---------------- link predictor (refactored concat GEMM) ----------------
    gemm(neb, Wl1b,            Abuf, N, BH, BH);   // A = emb @ Wl1[0:768,:]
    gemm(neb, Wl1b + BH * BH,  Bbuf, N, BH, BH);   // B = emb @ Wl1[768:1536,:]
    k_pair_predict<<<CDIV(P, 8), 256, 0, stream>>>(pe, P, Abuf, Bbuf, bl1, Wl2, bl2, out);
}
